// MSBlock_12962211299874
// MI455X (gfx1250) — matrix operand order repaired and checked by reading, not compiled, not run on hardware
//
#include <hip/hip_runtime.h>
#include <hip/hip_bf16.h>

typedef __attribute__((ext_vector_type(16))) __bf16 v16bf;
typedef __attribute__((ext_vector_type(8)))  float  v8f;

#define LSEQ 4096
#define CTOT 256
#define NB   16

static __device__ __forceinline__ unsigned short f2bf(float f) {
    union { float f; unsigned u; } v; v.f = f;
    unsigned r = v.u + 0x7FFFu + ((v.u >> 16) & 1u);
    return (unsigned short)(r >> 16);
}
static __device__ __forceinline__ float bf2f(unsigned short u) {
    union { unsigned u; float f; } v; v.u = ((unsigned)u) << 16; return v.f;
}

// ---------------- prep kernels ----------------

__global__ __launch_bounds__(256) void cvt_x_kernel(const float* __restrict__ x,
                                                    unsigned short* __restrict__ xbf) {
    int i = blockIdx.x * 256 + threadIdx.x;           // grid covers NB*CTOT*LSEQ
    xbf[i] = f2bf(x[i]);
}

// Fold BN (inference) into conv weights/bias; repack W to [branch][oc][k=t*64+ic] bf16.
__global__ __launch_bounds__(256) void prep_conv_kernel(
    const float* __restrict__ cw, const float* __restrict__ cb,
    const float* __restrict__ gamma, const float* __restrict__ beta,
    const float* __restrict__ mu, const float* __restrict__ var,
    unsigned short* __restrict__ Wc, float* __restrict__ biasc) {
    int tid = threadIdx.x;                 // one block of 256: branch*64+oc
    int branch = tid >> 6, oc = tid & 63;
    int ci = branch * 64 + oc;
    float inv = rsqrtf(var[ci] + 1e-5f);
    float sc  = gamma[ci] * inv;
    biasc[ci] = (cb[ci] - mu[ci]) * sc + beta[ci];
    for (int k = 0; k < 192; ++k) {
        int t = k >> 6, ic = k & 63;
        float wv = cw[(((size_t)branch * 64 + oc) * 64 + ic) * 3 + t] * sc;
        Wc[(size_t)ci * 192 + k] = f2bf(wv);
    }
}

// Effective fusion weights: prefix sums folded into W.  Weff[o][j], j<256 -> coeff on h,
// j>=256 -> coeff on l (= Wl' - Ws).
__global__ __launch_bounds__(256) void prep_fusion_kernel(const float* __restrict__ fw,
                                                          unsigned short* __restrict__ Weff) {
    int idx = blockIdx.x * 256 + threadIdx.x;     // 256*512 total
    int o = idx >> 9, j = idx & 511;
    float v;
    if (j < 256) {
        int i = j >> 6, ch = j & 63;
        float s = 0.f;
        for (int bb = i; bb < 4; ++bb) s += fw[(size_t)o * 512 + bb * 64 + ch];
        v = s;
    } else {
        int j2 = j - 256; int i = j2 >> 6, ch = j2 & 63;
        float sl = 0.f, ss = 0.f;
        for (int bb = 0; bb <= i; ++bb) sl += fw[(size_t)o * 512 + 256 + bb * 64 + ch];
        for (int bb = i; bb < 4; ++bb)  ss += fw[(size_t)o * 512 + bb * 64 + ch];
        v = sl - ss;
    }
    Weff[idx] = f2bf(v);
}

// ---------------- kernel 1: dilated conv + BN + GELU via WMMA ----------------
// GEMM per (batch, branch): M=64 oc, K=192 (=3 taps * 64 ic), N tile of 128 positions.
__global__ __launch_bounds__(256) void branch_conv_kernel(
    const unsigned short* __restrict__ xbf,   // [NB,256,L] bf16
    const unsigned short* __restrict__ Wc,    // [4,64,192] bf16
    const float* __restrict__ biasc,          // [256]
    unsigned short* __restrict__ hbuf) {      // [NB,256,L] bf16
    const int branch = blockIdx.z;
    const int b      = blockIdx.y;
    const int n0     = blockIdx.x * 128;
    const int d      = 1 << branch;
    const int tid  = threadIdx.x;
    const int lane = tid & 31;
    const int w    = tid >> 5;
    const int half = lane >> 4;

    __shared__ alignas(32) unsigned short sW[64 * 192];   // [m][k]
    __shared__ alignas(32) unsigned short sXT[144 * 64];  // [q][ic], q = pos - (n0 - d)

    for (int i = tid; i < 64 * 192; i += 256)
        sW[i] = Wc[(size_t)branch * 64 * 192 + i];

    const int span = 128 + 2 * d;
    const unsigned short* xsrc = xbf + ((size_t)b * CTOT + branch * 64) * LSEQ;
    for (int i = tid; i < span * 64; i += 256) {
        int q = i >> 6, ic = i & 63;
        int p = n0 - d + q;
        unsigned short v = 0;                 // conv zero padding
        if (p >= 0 && p < LSEQ) v = xsrc[(size_t)ic * LSEQ + p];
        sXT[q * 64 + ic] = v;
    }
    __syncthreads();

    const int mo    = (w & 3) * 16;
    const int nbase = (w >> 2) * 64;
    v8f zero = {0.f, 0.f, 0.f, 0.f, 0.f, 0.f, 0.f, 0.f};
    v8f acc[4] = {zero, zero, zero, zero};

    const int mA = mo + (lane & 15);
#pragma unroll
    for (int kk = 0; kk < 6; ++kk) {
        const int kbase = kk * 32;
        union { v16bf v; uint4 q[2]; } A;     // A 16x32 bf16, k = 16*half + s (contiguous, same as B)
        A.q[0] = *(const uint4*)&sW[mA * 192 + kbase + half * 16];
        A.q[1] = *(const uint4*)&sW[mA * 192 + kbase + half * 16 + 8];
        const int kb2 = kbase + half * 16;    // lane-half K base for B
        const int t   = kb2 >> 6;             // tap (constant within 16-run)
        const int icb = kb2 & 63;
#pragma unroll
        for (int s = 0; s < 4; ++s) {
            const int nl = nbase + s * 16 + (lane & 15);
            const int qq = nl + t * d;
            v16bf Bv = *(const v16bf*)&sXT[qq * 64 + icb];   // 32B aligned contiguous
            acc[s] = __builtin_amdgcn_wmma_f32_16x16x32_bf16(
                false, A.v, false, Bv, (short)0, acc[s], false, false);
        }
    }

    const int rowbase = mo + half * 8;
    const int colb    = lane & 15;
#pragma unroll
    for (int s = 0; s < 4; ++s) {
        int pos = n0 + nbase + s * 16 + colb;
#pragma unroll
        for (int r = 0; r < 8; ++r) {
            int row = rowbase + r;
            float vv = acc[s][r] + biasc[branch * 64 + row];
            float g  = 0.5f * vv * (1.0f + erff(vv * 0.70710678118654752f)); // exact GELU
            hbuf[((size_t)b * CTOT + branch * 64 + row) * LSEQ + pos] = f2bf(g);
        }
    }
}

// ---------------- kernel 2: moving average (window 25, edge pad) ----------------
__global__ __launch_bounds__(256) void mavg_kernel(const unsigned short* __restrict__ hbuf,
                                                   unsigned short* __restrict__ lbuf) {
    __shared__ float srow[LSEQ];
    const size_t base = (size_t)blockIdx.x * LSEQ;     // one (b,ch) row per WG
    const int tid = threadIdx.x;
    for (int i = tid; i < LSEQ; i += 256) srow[i] = bf2f(hbuf[base + i]);
    __syncthreads();
    const int p0 = tid * 16;
    float sum = 0.f;
#pragma unroll
    for (int j = -12; j <= 12; ++j) {
        int q = p0 + j; q = q < 0 ? 0 : (q > LSEQ - 1 ? LSEQ - 1 : q);
        sum += srow[q];
    }
    const float inv = 1.0f / 25.0f;
    lbuf[base + p0] = f2bf(sum * inv);
#pragma unroll
    for (int p = p0 + 1; p < p0 + 16; ++p) {
        int qa = p + 12; qa = qa > LSEQ - 1 ? LSEQ - 1 : qa;
        int qr = p - 13; qr = qr < 0 ? 0 : qr;
        sum += srow[qa] - srow[qr];
        lbuf[base + p] = f2bf(sum * inv);
    }
}

// ---------------- kernel 3: fusion GEMM + ReLU + residual via WMMA ----------------
// out[b,o,p] = relu( Weff[o,:] . [h;l][b,:,p] + fb[o] ) + alpha * x[b,o,p]
__global__ __launch_bounds__(256) void fusion_kernel(
    const unsigned short* __restrict__ hbuf, const unsigned short* __restrict__ lbuf,
    const unsigned short* __restrict__ Weff, const float* __restrict__ fb,
    const float* __restrict__ alpha, const float* __restrict__ x,
    float* __restrict__ out) {
    const int b  = blockIdx.z;
    const int Mb = blockIdx.y * 128;
    const int n0 = blockIdx.x * 128;
    const int tid  = threadIdx.x;
    const int lane = tid & 31;
    const int w    = tid >> 5;
    const int half = lane >> 4;
    const int wm = w & 3, wn = w >> 2;

    __shared__ alignas(32) unsigned short sA[128 * 32]; // [m][k]
    __shared__ alignas(32) unsigned short sB[128 * 32]; // [n][k]

    v8f zero = {0.f, 0.f, 0.f, 0.f, 0.f, 0.f, 0.f, 0.f};
    v8f acc[2][4] = {{zero, zero, zero, zero}, {zero, zero, zero, zero}};

    for (int kk = 0; kk < 16; ++kk) {
        const int kbase = kk * 32;
        __syncthreads();
        {   // A tile: 128x32 from Weff (row-major, aligned 16B chunks)
            int m = tid >> 1, ko = (tid & 1) * 16;
            const uint4* src = (const uint4*)&Weff[(size_t)(Mb + m) * 512 + kbase + ko];
            *(uint4*)&sA[m * 32 + ko]     = src[0];
            *(uint4*)&sA[m * 32 + ko + 8] = src[1];
        }
        {   // B tile: transpose 32(K) x 128(N) -> [n][k]
            int k = tid & 31, ng = tid >> 5;
            int j = kbase + k;
            const unsigned short* src = (j < 256)
                ? &hbuf[((size_t)b * CTOT + j) * LSEQ + n0 + ng * 16]
                : &lbuf[((size_t)b * CTOT + (j - 256)) * LSEQ + n0 + ng * 16];
            union { uint4 q[2]; unsigned short s[16]; } t;
            t.q[0] = ((const uint4*)src)[0];
            t.q[1] = ((const uint4*)src)[1];
#pragma unroll
            for (int e = 0; e < 16; ++e) sB[(ng * 16 + e) * 32 + k] = t.s[e];
        }
        __syncthreads();
#pragma unroll
        for (int ms = 0; ms < 2; ++ms) {
            int m = wm * 32 + ms * 16 + (lane & 15);
            union { v16bf v; uint4 q[2]; } A;
            A.q[0] = *(const uint4*)&sA[m * 32 + half * 16];
            A.q[1] = *(const uint4*)&sA[m * 32 + half * 16 + 8];
#pragma unroll
            for (int ns = 0; ns < 4; ++ns) {
                int n = wn * 64 + ns * 16 + (lane & 15);
                v16bf Bv = *(const v16bf*)&sB[n * 32 + half * 16];
                acc[ms][ns] = __builtin_amdgcn_wmma_f32_16x16x32_bf16(
                    false, A.v, false, Bv, (short)0, acc[ms][ns], false, false);
            }
        }
    }

    const float al = alpha[0];
#pragma unroll
    for (int ms = 0; ms < 2; ++ms) {
#pragma unroll
        for (int ns = 0; ns < 4; ++ns) {
            int pos = n0 + wn * 64 + ns * 16 + (lane & 15);
#pragma unroll
            for (int r = 0; r < 8; ++r) {
                int row = Mb + wm * 32 + ms * 16 + half * 8 + r;
                size_t oi = ((size_t)b * CTOT + row) * LSEQ + pos;
                float v = acc[ms][ns][r] + fb[row];
                v = v > 0.f ? v : 0.f;
                out[oi] = v + al * x[oi];
            }
        }
    }
}

// ---------------- launcher ----------------
extern "C" void kernel_launch(void* const* d_in, const int* in_sizes, int n_in,
                              void* d_out, int out_size, void* d_ws, size_t ws_size,
                              hipStream_t stream) {
    (void)in_sizes; (void)n_in; (void)out_size; (void)ws_size;
    const float* x     = (const float*)d_in[0];
    const float* cw    = (const float*)d_in[1];
    const float* cb    = (const float*)d_in[2];
    const float* gamma = (const float*)d_in[3];
    const float* beta  = (const float*)d_in[4];
    const float* mu    = (const float*)d_in[5];
    const float* var   = (const float*)d_in[6];
    const float* fw    = (const float*)d_in[7];
    const float* fbias = (const float*)d_in[8];
    const float* alpha = (const float*)d_in[9];
    float* out = (float*)d_out;

    const size_t NX = (size_t)NB * CTOT * LSEQ;          // 16,777,216
    char* ws = (char*)d_ws;
    unsigned short* xbf   = (unsigned short*)(ws);                     // 32 MB
    unsigned short* hbuf  = (unsigned short*)(ws + NX * 2);            // 32 MB
    unsigned short* lbuf  = (unsigned short*)(ws + NX * 4);            // 32 MB
    unsigned short* Wc    = (unsigned short*)(ws + NX * 6);            // 96 KB
    float*          biasc = (float*)(ws + NX * 6 + 4 * 64 * 192 * 2);
    unsigned short* Weff  = (unsigned short*)(ws + NX * 6 + 4 * 64 * 192 * 2 + 256 * 4);

    // prep
    cvt_x_kernel<<<dim3((unsigned)(NX / 256)), dim3(256), 0, stream>>>(x, xbf);
    prep_conv_kernel<<<dim3(1), dim3(256), 0, stream>>>(cw, cb, gamma, beta, mu, var, Wc, biasc);
    prep_fusion_kernel<<<dim3(512), dim3(256), 0, stream>>>(fw, Weff);
    // branch conv + BN + GELU (WMMA)
    branch_conv_kernel<<<dim3(LSEQ / 128, NB, 4), dim3(256), 0, stream>>>(xbf, Wc, biasc, hbuf);
    // moving-average trend
    mavg_kernel<<<dim3(NB * CTOT), dim3(256), 0, stream>>>(hbuf, lbuf);
    // fusion GEMM + ReLU + residual (WMMA)
    fusion_kernel<<<dim3(LSEQ / 128, CTOT / 128, NB), dim3(256), 0, stream>>>(
        hbuf, lbuf, Weff, fbias, alpha, x, out);
}
